// Model_16140487098552
// MI455X (gfx1250) — matrix operand order repaired and checked by reading, not compiled, not run on hardware
//
#include <hip/hip_runtime.h>
#include <hip/hip_bf16.h>

typedef __attribute__((ext_vector_type(16))) __bf16 v16bf;
typedef __attribute__((ext_vector_type(8)))  float  v8f;
typedef unsigned short ushort_t;

#define DIMK       1024
#define VOCAB      50257
#define ROWS       64                    // tokens per workgroup (4 x 16-row WMMA subtiles)
#define SUBT       4
#define NTILES     ((VOCAB + 15) / 16)   // 3142 vocab column tiles
#define LDS_STRIDE 1032                  // halves; 1024 + 8 pad -> conflict-free b128 row reads
#define SOFTCAP_F  30.0f

union V16 {
    uint4 u[2];
    v16bf v;
};

// exp(softcap(L) - 30) with softcap(L) = 30*tanh(L/30).
// tanh(L/30) = (e-1)/(e+1), e = exp(L/15)  =>  softcap(L)-30 = -60/(e+1).
__device__ __forceinline__ float softcap_expm30(float L) {
    float e = __expf(L * (1.0f / 15.0f));
    return __expf(-60.0f / (e + 1.0f));
}

__device__ __forceinline__ float bf_lo(unsigned u) {
    return __builtin_bit_cast(float, u << 16);
}
__device__ __forceinline__ float bf_hi(unsigned u) {
    return __builtin_bit_cast(float, u & 0xFFFF0000u);
}

// ---------------------------------------------------------------------------
// Kernel 1: partial exp-sums of softcapped logits.
// psum[split][token] = sum_{v in split} exp(softcap(x_t . W_v) - 30)
// Grid: (n_tok/ROWS, nsplit). Each wave streams a 1/8 stripe of its split's
// vocab tiles; one weight B-fragment feeds 4 WMMAs (4 row sub-tiles).
// ---------------------------------------------------------------------------
__global__ __launch_bounds__(256) void ce_partial_kernel(
    const ushort_t* __restrict__ x,      // [n_tok][1024] bf16 bits
    const ushort_t* __restrict__ w,      // [VOCAB][1024] bf16 bits
    float* __restrict__ psum,            // [nsplit][n_tok]
    int n_tok, int nsplit)
{
    extern __shared__ __align__(16) unsigned char smem_raw[];
    ushort_t* sx = (ushort_t*)smem_raw;                              // ROWS x LDS_STRIDE halves
    float* part  = (float*)(smem_raw + ROWS * LDS_STRIDE * 2);       // [8 waves][ROWS]

    const int tid   = threadIdx.x;
    const int wave  = tid >> 5;
    const int lane  = tid & 31;
    const int l15   = lane & 15;
    const int half  = lane >> 4;
    const int tok0  = blockIdx.x * ROWS;
    const int split = blockIdx.y;

    // ---- stage 64x1024 bf16 x-tile into LDS (padded row stride) ----
    {
        const uint4* xg = (const uint4*)(x + (size_t)tok0 * DIMK);
        #pragma unroll 4
        for (int i = tid; i < ROWS * (DIMK / 8); i += 256) {   // 8192 16B chunks
            int r = i >> 7;          // 128 chunks per row
            int c = i & 127;
            uint4 v = xg[(size_t)r * (DIMK / 8) + c];
            *(uint4*)(sx + r * LDS_STRIDE + c * 8) = v;
        }
    }
    __syncthreads();

    const int klo = half * 8;    // A-fragment K start (halves), per CDNA5 ISA 16-bit A layout
    const int kb  = half * 8;    // B-fragment K start, same table order as A

    // Loop-invariant LDS byte offsets for the 4 A sub-tile fragment bases.
    // The empty asm keeps each as its own VGPR base register so DS offsets
    // stay small immediates (kk*2 + {0,32}) instead of >64KB folds that force
    // in-loop v_adds (and WMMA->VALU hazard v_nops).
    unsigned aoff[SUBT];
    #pragma unroll
    for (int s = 0; s < SUBT; ++s) {
        unsigned o = (unsigned)(((s * 16 + l15) * LDS_STRIDE + klo) * 2);
        asm("" : "+v"(o));
        aoff[s] = o;
    }

    float rs[SUBT][8];
    #pragma unroll
    for (int s = 0; s < SUBT; ++s)
        #pragma unroll
        for (int j = 0; j < 8; ++j) rs[s][j] = 0.0f;

    const int t0 = (int)(((long long)NTILES * split) / nsplit);
    const int t1 = (int)(((long long)NTILES * (split + 1)) / nsplit);

    // uniform per-wave loop: EXEC all-1s around the WMMAs
    for (int t = t0 + wave; t < t1; t += 8) {
        const int col = t * 16 + l15;
        const int ncl = (col < VOCAB) ? col : (VOCAB - 1);     // clamp tail-tile loads
        const ushort_t* wr = w + (size_t)ncl * DIMK + kb;

        // prefetch this stripe's next tile weight row
        {
            int cn = col + 128;
            const ushort_t* pn = w + (size_t)((cn < VOCAB) ? cn : (VOCAB - 1)) * DIMK + kb;
            __builtin_prefetch(pn, 0, 1);
        }

        v8f c[SUBT];
        #pragma unroll
        for (int s = 0; s < SUBT; ++s) c[s] = (v8f){};

        #pragma unroll 2
        for (int kk = 0; kk < DIMK; kk += 32) {
            // B fragment: halves {8h..8h+7, 16+8h..16+8h+7} of one (clamped) weight row
            V16 b;
            b.u[0] = *(const uint4*)(wr + kk);
            b.u[1] = *(const uint4*)(wr + kk + 16);

            // All 4 A fragments up front -> one DS clause, staged waits,
            // then 4 back-to-back WMMAs (no cross-WMMA register deps).
            V16 a[SUBT];
            #pragma unroll
            for (int s = 0; s < SUBT; ++s) {
                const ushort_t* sp = (const ushort_t*)(smem_raw + aoff[s]) + kk;
                a[s].u[0] = *(const uint4*)(sp);
                a[s].u[1] = *(const uint4*)(sp + 16);
            }

            #pragma unroll
            for (int s = 0; s < SUBT; ++s)
                c[s] = __builtin_amdgcn_wmma_f32_16x16x32_bf16(
                           false, a[s].v, false, b.v, (short)0, c[s], false, false);
        }

        if (col < VOCAB) {   // divergence only after the WMMAs for this tile
            #pragma unroll
            for (int s = 0; s < SUBT; ++s)
                #pragma unroll
                for (int j = 0; j < 8; ++j)
                    rs[s][j] += softcap_expm30(c[s][j]);
        }
    }

    // ---- reduce partial sums across the 16 lanes that share each row ----
    #pragma unroll
    for (int s = 0; s < SUBT; ++s) {
        #pragma unroll
        for (int j = 0; j < 8; ++j) {
            float v = rs[s][j];
            v += __shfl_xor(v, 1, 32);
            v += __shfl_xor(v, 2, 32);
            v += __shfl_xor(v, 4, 32);
            v += __shfl_xor(v, 8, 32);
            rs[s][j] = v;
        }
    }
    if (l15 == 0) {
        // C/D layout: VGPR j in half h belongs to row (sub*16 + 8*h + j)
        #pragma unroll
        for (int s = 0; s < SUBT; ++s)
            #pragma unroll
            for (int j = 0; j < 8; ++j)
                part[wave * ROWS + s * 16 + half * 8 + j] = rs[s][j];
    }
    __syncthreads();

    // ---- combine across the 8 waves; write the split's partial sum ----
    if (tid < ROWS) {
        float tot = 0.0f;
        #pragma unroll
        for (int wv = 0; wv < 8; ++wv) tot += part[wv * ROWS + tid];
        psum[(size_t)split * n_tok + tok0 + tid] = tot;
    }
}

// ---------------------------------------------------------------------------
// Kernel 2: finish. out[token] = 30 + log(sum_s psum[s][token])
//                               - softcap( x_t . W[target_t] )
// One wave32 per token; lane k handles 32 contiguous halves of the dot.
// ---------------------------------------------------------------------------
__global__ __launch_bounds__(256) void ce_finish_kernel(
    const ushort_t* __restrict__ x,
    const int* __restrict__ targets,
    const ushort_t* __restrict__ w,
    const float* __restrict__ psum,
    float* __restrict__ out,
    int n_tok, int nsplit)
{
    const int wave  = threadIdx.x >> 5;
    const int lane  = threadIdx.x & 31;
    const int token = blockIdx.x * 8 + wave;
    if (token >= n_tok) return;

    int tv = targets[token];
    if (tv < 0) tv = 0;
    if (tv >= VOCAB) tv = VOCAB - 1;

    const uint4* xr = (const uint4*)(x + (size_t)token * DIMK);
    const uint4* wr = (const uint4*)(w + (size_t)tv * DIMK);

    float acc = 0.0f;
    #pragma unroll
    for (int u = 0; u < 4; ++u) {            // 4 * 8 halves = 32 halves per lane
        uint4 a = xr[lane * 4 + u];
        uint4 b = wr[lane * 4 + u];
        acc = fmaf(bf_lo(a.x), bf_lo(b.x), acc);
        acc = fmaf(bf_hi(a.x), bf_hi(b.x), acc);
        acc = fmaf(bf_lo(a.y), bf_lo(b.y), acc);
        acc = fmaf(bf_hi(a.y), bf_hi(b.y), acc);
        acc = fmaf(bf_lo(a.z), bf_lo(b.z), acc);
        acc = fmaf(bf_hi(a.z), bf_hi(b.z), acc);
        acc = fmaf(bf_lo(a.w), bf_lo(b.w), acc);
        acc = fmaf(bf_hi(a.w), bf_hi(b.w), acc);
    }
    acc += __shfl_xor(acc, 16, 32);
    acc += __shfl_xor(acc, 8, 32);
    acc += __shfl_xor(acc, 4, 32);
    acc += __shfl_xor(acc, 2, 32);
    acc += __shfl_xor(acc, 1, 32);

    if (lane == 0) {
        float tot = 0.0f;
        for (int s = 0; s < nsplit; ++s) tot += psum[(size_t)s * n_tok + token];
        float e = __expf(acc * (1.0f / 15.0f));
        float z = SOFTCAP_F * (e - 1.0f) / (e + 1.0f);   // 30*tanh(acc/30)
        out[token] = SOFTCAP_F + __logf(tot) - z;
    }
}

// ---------------------------------------------------------------------------
extern "C" void kernel_launch(void* const* d_in, const int* in_sizes, int n_in,
                              void* d_out, int out_size, void* d_ws, size_t ws_size,
                              hipStream_t stream) {
    (void)n_in; (void)out_size;
    const ushort_t* x  = (const ushort_t*)d_in[0];   // bf16 [B,S,D]
    const int* targets = (const int*)d_in[1];        // int  [B,S]
    const ushort_t* wt = (const ushort_t*)d_in[2];   // bf16 [V,D]
    float* out = (float*)d_out;                      // f32  [B,S]

    const int n_tok = in_sizes[0] / DIMK;            // 8192

    // 4-way vocab split for parallelism if scratch allows; else stage in d_out.
    int nsplit;
    float* psum;
    if (ws_size >= (size_t)4 * n_tok * sizeof(float)) {
        nsplit = 4;
        psum = (float*)d_ws;
    } else {
        nsplit = 1;
        psum = out;
    }

    const size_t lds_bytes = (size_t)ROWS * LDS_STRIDE * 2 + 8 * ROWS * sizeof(float);
    dim3 grid1(n_tok / ROWS, nsplit);
    ce_partial_kernel<<<grid1, 256, lds_bytes, stream>>>(x, wt, psum, n_tok, nsplit);
    ce_finish_kernel<<<(n_tok + 7) / 8, 256, 0, stream>>>(x, targets, wt, psum, out, n_tok, nsplit);
}
